// GraphFlow_89051851915597
// MI455X (gfx1250) — matrix operand order repaired and checked by reading, not compiled, not run on hardware
//
#include <hip/hip_runtime.h>
#include <hip/hip_bf16.h>

// ---------------- common types / constants ----------------
typedef __bf16 bf16;
typedef __attribute__((ext_vector_type(16))) __bf16 v16bf;
typedef __attribute__((ext_vector_type(8)))  float  v8f;

#define NB 32      // batch
#define NT 32      // sequence length == n_atoms
#define NH 128     // GRU hidden
#define G3 384     // 3*NH
#define NGCU 64
#define NDENSE 64
#define CATD 576   // 64 + 4*128
#define D0IN 704   // 128 + 576

// ---------------- bf16 helper (bit-exact RNE, storage-only __bf16) --------
__device__ __host__ inline bf16 f2bf(float f) {
  union { float f; unsigned u; } v; v.f = f;
  unsigned r = (v.u + 0x7FFFu + ((v.u >> 16) & 1u)) >> 16;
  union { unsigned short s; bf16 b; } o; o.s = (unsigned short)r;
  return o.b;
}

__device__ inline float sigm(float x) { return 1.f / (1.f + __expf(-x)); }

// ------------- WMMA fragment layout helpers (CDNA5 ISA 7.12.2) -----------
// A (16x32 bf16): lane = row + half*16 ; element e encodes K per ISA table.
__device__ inline void a_slot(int r, int k, int& lane, int& e) {
  int blk = k >> 4, rem = k & 15, half = rem >> 3, rem2 = rem & 7;
  int within = rem2 >> 1, slot = rem2 & 1;
  lane = r + (half << 4);
  e = (((blk << 2) | within) << 1) | slot;
}

// store one f32 value into a bf16 A-fragment buffer laid out
// [mt*Ktiles + kt][lane][e] for a (16*Mtiles) x (32*Ktiles) matrix
__device__ inline void store_a_frag(bf16* buf, int Ktiles, int row, int k, float val) {
  int mt = row >> 4, kt = k >> 5;
  int lane, e; a_slot(row & 15, k & 31, lane, e);
  buf[((((mt * Ktiles + kt) << 5) + lane) << 4) + e] = f2bf(val);
}

// generic tiled matmul: out(16*Mtiles x 16*Ntiles) = Afrag * Bswz, K = 32*Ktiles
// Bswz layout: [kt*Ntiles + nt][lane][e] with k = (e>>3)*16 + (lane>>4)*8 + (e&7), n = lane&15
__device__ inline void wmma_mm(const bf16* afrag, const bf16* bswz, float* out,
                               int Mtiles, int Ntiles, int Ktiles) {
  const int nw   = blockDim.x >> 5;
  const int wv   = threadIdx.x >> 5;
  const int lane = threadIdx.x & 31;
  const int ld   = Ntiles << 4;
  for (int tile = wv; tile < Mtiles * Ntiles; tile += nw) {
    int mt = tile / Ntiles, nt = tile % Ntiles;
    v8f acc = {};
    for (int kt = 0; kt < Ktiles; ++kt) {
      const v16bf* ap = (const v16bf*)(afrag + ((((mt * Ktiles + kt) << 5) + lane) << 4));
      const v16bf* bp = (const v16bf*)(bswz  + ((((kt * Ntiles + nt) << 5) + lane) << 4));
      acc = __builtin_amdgcn_wmma_f32_16x16x32_bf16(false, *ap, false, *bp,
                                                    (short)0, acc, false, false);
    }
    float* dst = out + (mt << 4) * ld + (nt << 4);
    int n = lane & 15, mh = lane >> 4;
#pragma unroll
    for (int v = 0; v < 8; ++v) dst[((mh << 3) + v) * ld + n] = acc[v];
  }
}

// shared GRU gate math (Keras reset_after=True, gates [z,r,h])
__device__ inline void gru_gates(float* h, bf16* hfrag,
                                 const float* xw, int xw_bstride,
                                 const float* rec,
                                 const float* bi, const float* bh,
                                 float* out, int out_bstride) {
  for (int i = threadIdx.x; i < NB * NH; i += blockDim.x) {
    int b = i >> 7, j = i & 127;
    const float* xrow = xw + b * xw_bstride;
    const float* rrow = rec + b * G3;
    float xz = xrow[j]        + bi[j];
    float xr = xrow[NH + j]   + bi[NH + j];
    float xh = xrow[2*NH + j] + bi[2*NH + j];
    float rz = rrow[j]        + bh[j];
    float rr = rrow[NH + j]   + bh[NH + j];
    float rh = rrow[2*NH + j] + bh[2*NH + j];
    float z  = sigm(xz + rz);
    float r  = sigm(xr + rr);
    float hh = tanhf(xh + r * rh);
    float hn = z * h[i] + (1.f - z) * hh;
    h[i] = hn;
    store_a_frag(hfrag, 4, b, j, hn);
    if (out) out[b * out_bstride + j] = hn;
  }
}

// ---------------- kernel: weight swizzle f32 -> bf16 B-fragments ---------
__global__ void k_swizzle(const float* __restrict__ src, bf16* __restrict__ dst,
                          int K, int N) {
  int total = K * N;
  int Ntiles = N >> 4;
  for (int i = blockIdx.x * blockDim.x + threadIdx.x; i < total;
       i += gridDim.x * blockDim.x) {
    int k = i / N, n = i % N;
    int kt = k >> 5, nt = n >> 4;
    int lane = (((k >> 3) & 1) << 4) | (n & 15);
    int e = (((k >> 4) & 1) << 3) | (k & 7);
    dst[((((kt * Ntiles + nt) << 5) + lane) << 4) + e] = f2bf(src[i]);
  }
}

// ---------------- kernel: graph conv + gather ----------------------------
__global__ void k_graph(const float* __restrict__ atoms, const float* __restrict__ adj,
                        const int* __restrict__ walk,
                        const float* __restrict__ gcW, const float* __restrict__ gcb,
                        float* __restrict__ hw_out) {
  __shared__ float dsh[32 * 64];
  __shared__ float adj2[32 * 32];
  __shared__ float hg[32 * 64];
  int tid = threadIdx.x;
  for (int i = tid; i < 32 * 64; i += blockDim.x) {
    int a = i >> 6, o = i & 63;
    float s = gcb[o];
    for (int k = 0; k < 16; ++k) s += atoms[a * 16 + k] * gcW[k * 64 + o];
    dsh[i] = s;
  }
  for (int i = tid; i < 32 * 32; i += blockDim.x) {
    int r = i >> 5, c = i & 31;
    adj2[i] = adj[r * 32 + c] + adj[c * 32 + r] + (r == c ? 1.f : 0.f);
  }
  __syncthreads();
  for (int i = tid; i < 32 * 64; i += blockDim.x) {
    int a = i >> 6, o = i & 63;
    float s = 0.f;
    for (int j = 0; j < 32; ++j) s += adj2[a * 32 + j] * dsh[j * 64 + o];
    hg[i] = tanhf(s);
  }
  __syncthreads();
  for (int i = tid; i < NB * NT * 64; i += blockDim.x) {
    int bt = i >> 6, o = i & 63;
    hw_out[i] = hg[walk[bt] * 64 + o];
  }
}

// ---------------- kernel: one full GRU layer over the sequence -----------
// single workgroup (8 waves). Phase A: xw = X@W (WMMA). Phase B: recurrence.
__global__ void k_gru_layer(const float* __restrict__ xseq, int in_dim,
                            const bf16* __restrict__ Wswz, const bf16* __restrict__ Uswz,
                            const float* __restrict__ bi, const float* __restrict__ bh,
                            float* __restrict__ xw, float* __restrict__ out_seq,
                            float* __restrict__ last_state, int reverse) {
  extern __shared__ unsigned char smem[];
  float* h     = (float*)smem;                               // 16384 B
  bf16*  hfrag = (bf16*)(smem + 16384);                      //  8192 B
  float* rec   = (float*)(smem + 16384 + 8192);              // 49152 B
  bf16*  astg  = (bf16*)(smem + 16384 + 8192 + 49152);       //  4096 B
  int tid = threadIdx.x;
  int Kt = in_dim >> 5;

  // pull recurrent weights toward the WGP before the sequential loop
  for (int i = tid * 64; i < 96 * 512 * 2; i += blockDim.x * 64)
    __builtin_prefetch(((const char*)Uswz) + i, 0, 3);

  // ---- Phase A: xw[s][n] = x[s] @ W, s in processed-time order ----
  for (int mt = 0; mt < (NB * NT) / 16; ++mt) {
    for (int i = tid; i < 16 * in_dim; i += blockDim.x) {
      int r = i / in_dim, k = i % in_dim;
      int s = (mt << 4) + r;
      int b = s >> 5, t0 = s & 31;
      int tt = reverse ? (NT - 1 - t0) : t0;
      store_a_frag(astg, Kt, r, k, xseq[(b * NT + tt) * in_dim + k]);
    }
    __syncthreads();
    wmma_mm(astg, Wswz, xw + mt * 16 * G3, 1, 24, Kt);
    __syncthreads();
  }
  // ---- Phase B: recurrence ----
  for (int i = tid; i < NB * NH; i += blockDim.x) {
    h[i] = 0.f;
    store_a_frag(hfrag, 4, i >> 7, i & 127, 0.f);
  }
  __threadfence();
  __syncthreads();
  for (int t0 = 0; t0 < NT; ++t0) {
    wmma_mm(hfrag, Uswz, rec, 2, 24, 4);
    __syncthreads();
    gru_gates(h, hfrag, xw + t0 * G3, NT * G3, rec, bi, bh,
              out_seq + t0 * NH, NT * NH);
    __syncthreads();
  }
  if (last_state)
    for (int i = tid; i < NB * NH; i += blockDim.x) last_state[i] = h[i];
}

// ---------------- autoregressive tail --------------------------------
struct ARArgs {
  const int*   walk;
  const float* noise2;
  const float* noise3;
  const float* hw; const float* fseq; const float* bseq;
  const float* fstate; const float* bstate;
  const float* W0raw; const bf16* U0swz; const float* bi0; const float* bh0;
  const bf16* W1swz;  const bf16* U1swz; const float* bi1; const float* bh1;
  const bf16* W2swz;  const bf16* U2swz; const float* bi2; const float* bh2;
  const bf16* d0Wswz; const float* d0b;
  const float* d1W; const float* d1b;
  const float* d2W; const float* d2b;
  const float* dw1W; const float* dw1b; const float* db1W; const float* db1b;
  const float* dw2W; const float* dw2b; const float* db2W; const float* db2b;
  const float* dw3W; const float* dw3b; const float* db3W; const float* db3b;
  float* out;
};

// dynamic-LDS layout (bytes)
#define AR_OFF_H0    0
#define AR_OFF_H1    16384
#define AR_OFF_H2    32768
#define AR_OFF_XW    49152
#define AR_OFF_REC   98304
#define AR_OFF_HF0   147456
#define AR_OFF_HF1   155648
#define AR_OFF_HF2   163840
#define AR_OFF_DIN   172032
#define AR_OFF_D0O   217088
#define AR_OFF_D1O   225280
#define AR_OFF_WFL   233472
#define AR_OFF_BFL   238080
#define AR_OFF_SEQ   239616
#define AR_OFF_XOUT  251904
#define AR_OFF_LGD   264192
#define AR_OFF_RED   264320
#define AR_OFF_ISN   264576
#define AR_SMEM      268672

__device__ inline void head_mm(const float* hin, const float* W, const float* bias,
                               float* out, int O) {
  for (int i = threadIdx.x; i < NB * O; i += blockDim.x) {
    int b = i / O, o = i % O;
    float s = bias[o];
    for (int k = 0; k < NDENSE; ++k) s += hin[b * NDENSE + k] * W[k * O + o];
    out[i] = s;
  }
}

__global__ void k_autoreg(ARArgs A) {
  extern __shared__ unsigned char smem[];
  float* h0     = (float*)(smem + AR_OFF_H0);
  float* h1     = (float*)(smem + AR_OFF_H1);
  float* h2     = (float*)(smem + AR_OFF_H2);
  float* xw     = (float*)(smem + AR_OFF_XW);
  float* rec    = (float*)(smem + AR_OFF_REC);
  bf16*  hf0    = (bf16*)(smem + AR_OFF_HF0);
  bf16*  hf1    = (bf16*)(smem + AR_OFF_HF1);
  bf16*  hf2    = (bf16*)(smem + AR_OFF_HF2);
  bf16*  dinf   = (bf16*)(smem + AR_OFF_DIN);
  float* d0o    = (float*)(smem + AR_OFF_D0O);
  float* d1o    = (float*)(smem + AR_OFF_D1O);
  float* wfl    = (float*)(smem + AR_OFF_WFL);
  float* bfl    = (float*)(smem + AR_OFF_BFL);
  float* seqbuf = (float*)(smem + AR_OFF_SEQ);
  float* xout   = (float*)(smem + AR_OFF_XOUT);
  float* logdet = (float*)(smem + AR_OFF_LGD);
  float* red    = (float*)(smem + AR_OFF_RED);
  int*   isnew  = (int*)  (smem + AR_OFF_ISN);
  int tid = threadIdx.x;

  // ---- init ----
  for (int i = tid; i < 3 * NB * NH; i += blockDim.x) h0[i] = 0.f;  // h0,h1,h2
  for (int i = tid; i < NB * NH; i += blockDim.x) {
    int b = i >> 7, j = i & 127;
    store_a_frag(hf0, 4, b, j, 0.f);
    store_a_frag(hf1, 4, b, j, 0.f);
    store_a_frag(hf2, 4, b, j, 0.f);
  }
  for (int i = tid; i < NB * NT * 3; i += blockDim.x) { seqbuf[i] = 0.f; xout[i] = 0.f; }
  if (tid < NB) logdet[tid] = 0.f;
  for (int i = tid; i < NB * NT; i += blockDim.x) {
    int b = i >> 5, t = i & 31;
    int v = A.walk[b * NT + t], nw = 1;
    for (int s = 0; s < t; ++s) if (A.walk[b * NT + s] == v) nw = 0;
    isnew[i] = nw;
  }
  __syncthreads();

  for (int p = 0; p < NT - 1; ++p) {
    int t = p + 1;
    // ---- xyz layer 0 (in=3, scalar xw) ----
    for (int i = tid; i < NB * G3; i += blockDim.x) {
      int b = i / G3, n = i % G3;
      const float* xin = seqbuf + (b * NT + p) * 3;
      xw[i] = xin[0] * A.W0raw[n] + xin[1] * A.W0raw[G3 + n] + xin[2] * A.W0raw[2 * G3 + n];
    }
    wmma_mm(hf0, A.U0swz, rec, 2, 24, 4);
    __syncthreads();
    gru_gates(h0, hf0, xw, G3, rec, A.bi0, A.bh0, nullptr, 0);
    __syncthreads();
    // ---- xyz layer 1 ----
    wmma_mm(hf0, A.W1swz, xw, 2, 24, 4);
    wmma_mm(hf1, A.U1swz, rec, 2, 24, 4);
    __syncthreads();
    gru_gates(h1, hf1, xw, G3, rec, A.bi1, A.bh1, nullptr, 0);
    __syncthreads();
    // ---- xyz layer 2 ----
    wmma_mm(hf1, A.W2swz, xw, 2, 24, 4);
    wmma_mm(hf2, A.U2swz, rec, 2, 24, 4);
    __syncthreads();
    gru_gates(h2, hf2, xw, G3, rec, A.bi2, A.bh2, nullptr, 0);
    __syncthreads();

    // ---- dense path: d0 via WMMA over [h_xyz | h_cat[b, idx]] ----
    for (int i = tid; i < NB * D0IN; i += blockDim.x) {
      int b = i / D0IN, j = i % D0IN;
      int idx = A.walk[b * NT + t];
      float v;
      if (j < NH) v = h2[b * NH + j];
      else {
        int q = j - NH;
        if      (q < 64)  v = A.hw[(b * NT + idx) * 64 + q];
        else if (q < 192) v = A.fseq[(b * NT + idx) * NH + (q - 64)];
        else if (q < 320) v = A.bseq[(b * NT + idx) * NH + (q - 192)];
        else if (q < 448) v = A.fstate[b * NH + (q - 320)];
        else              v = A.bstate[b * NH + (q - 448)];
      }
      store_a_frag(dinf, 22, b, j, v);
    }
    __syncthreads();
    wmma_mm(dinf, A.d0Wswz, d0o, 2, 4, 22);
    __syncthreads();
    for (int i = tid; i < NB * NDENSE; i += blockDim.x)
      d0o[i] = tanhf(d0o[i] + A.d0b[i & 63]);
    __syncthreads();
    for (int i = tid; i < NB * NDENSE; i += blockDim.x) {
      int b = i >> 6, o = i & 63;
      float s = A.d1b[o];
      for (int k = 0; k < NDENSE; ++k) s += d0o[b * 64 + k] * A.d1W[k * 64 + o];
      d1o[i] = tanhf(s);
    }
    __syncthreads();
    for (int i = tid; i < NB * NDENSE; i += blockDim.x) {
      int b = i >> 6, o = i & 63;
      float s = A.d2b[o];
      for (int k = 0; k < NDENSE; ++k) s += d1o[b * 64 + k] * A.d2W[k * 64 + o];
      d0o[i] = tanhf(s);  // reuse d0o as final dense output
    }
    __syncthreads();
    const float* hfin = d0o;

    // ---- flow heads ----
    if (p == 0) {
      head_mm(hfin, A.dw1W, A.dw1b, wfl, 4);
      head_mm(hfin, A.db1W, A.db1b, bfl, 4);
      __syncthreads();
      if (tid == 0) {
        float s = 0.f;
        for (int i = 0; i < NB * 4; ++i) s += wfl[i];
        red[0] = s;
      }
      __syncthreads();
      if (tid < NB) {
        int b = tid;
        float z = wfl[b*4] + wfl[b*4+1] + wfl[b*4+2] + wfl[b*4+3];
        for (int i = 0; i < 4; ++i) z = __expf(wfl[b*4+i]) * z + bfl[b*4+i];
        int idx = A.walk[b * NT + 1];
        float* xo = xout + (b * NT + idx) * 3;
        xo[0] = 0.f; xo[1] = 0.f; xo[2] = z;
        float* sb = seqbuf + (b * NT + 1) * 3;
        sb[0] = 0.f; sb[1] = 0.f; sb[2] = z;
        logdet[b] += red[0];
      }
    } else if (p == 1) {
      head_mm(hfin, A.dw2W, A.dw2b, wfl, 16);
      head_mm(hfin, A.db2W, A.db2b, bfl, 8);
      __syncthreads();
      if (tid < NB) {
        int b = tid;
        float z0 = A.noise2[b*2], z1 = A.noise2[b*2+1], dld = 0.f;
        for (int i = 0; i < 4; ++i) {
          const float* w = wfl + b*16 + i*4;             // 2x2 row-major
          float d0_ = w[0], d1_ = w[3];
          dld += d0_ + d1_;
          float e0 = __expf(d0_), e1 = __expf(d1_);
          float u01 = w[1], l10 = w[2];
          float W00 = e0,        W01 = e0*u01;
          float W10 = l10*e0,    W11 = l10*e0*u01 + e1;
          float n0 = z0*W00 + z1*W10 + bfl[b*8+i*2];
          float n1 = z0*W01 + z1*W11 + bfl[b*8+i*2+1];
          z0 = n0; z1 = n1;
        }
        const float* pv = seqbuf + (b * NT + 1) * 3;
        float x0 = pv[0], x1 = pv[1] + z0, x2 = pv[2] + z1;
        int idx = A.walk[b * NT + 2];
        float* xo = xout + (b * NT + idx) * 3;
        xo[0] = x0; xo[1] = x1; xo[2] = x2;
        float* sb = seqbuf + (b * NT + 2) * 3;
        sb[0] = x0; sb[1] = x1; sb[2] = x2;
        logdet[b] += dld;
      }
    } else {
      head_mm(hfin, A.dw3W, A.dw3b, wfl, 36);
      head_mm(hfin, A.db3W, A.db3b, bfl, 12);
      __syncthreads();
      if (tid < NB) {
        int b = tid;
        const float* nz = A.noise3 + (b * (NT - 3) + (t - 3)) * 3;
        float z0 = nz[0], z1 = nz[1], z2 = nz[2], dld = 0.f;
        for (int i = 0; i < 4; ++i) {
          const float* w = wfl + b*36 + i*9;             // 3x3 row-major
          float d0_ = w[0], d1_ = w[4], d2_ = w[8];
          dld += d0_ + d1_ + d2_;
          float e0 = __expf(d0_), e1 = __expf(d1_), e2 = __expf(d2_);
          float u01 = w[1], u02 = w[2], u12 = w[5];
          float l10 = w[3], l20 = w[6], l21 = w[7];
          float W00 = e0,     W01 = e0*u01,             W02 = e0*u02;
          float W10 = l10*e0, W11 = l10*e0*u01 + e1,    W12 = l10*e0*u02 + e1*u12;
          float W20 = l20*e0, W21 = l20*e0*u01 + l21*e1,W22 = l20*e0*u02 + l21*e1*u12 + e2;
          float n0 = z0*W00 + z1*W10 + z2*W20 + bfl[b*12+i*3];
          float n1 = z0*W01 + z1*W11 + z2*W21 + bfl[b*12+i*3+1];
          float n2 = z0*W02 + z1*W12 + z2*W22 + bfl[b*12+i*3+2];
          z0 = n0; z1 = n1; z2 = n2;
        }
        int idx = A.walk[b * NT + t];
        const float* pv = seqbuf + (b * NT + (t - 1)) * 3;
        float nx, ny, nzv;
        if (isnew[b * NT + t]) {
          nx = z0 + pv[0]; ny = z1 + pv[1]; nzv = z2 + pv[2];
          logdet[b] += dld;
        } else {
          const float* xo0 = xout + (b * NT + idx) * 3;
          nx = xo0[0]; ny = xo0[1]; nzv = xo0[2];
        }
        float* xo = xout + (b * NT + idx) * 3;
        xo[0] = nx; xo[1] = ny; xo[2] = nzv;
        float* sb = seqbuf + (b * NT + t) * 3;
        sb[0] = nx; sb[1] = ny; sb[2] = nzv;
      }
    }
    __syncthreads();
  }

  for (int i = tid; i < NB * NT * 3; i += blockDim.x) A.out[i] = xout[i];
  if (tid < NB) A.out[NB * NT * 3 + tid] = logdet[tid];
}

// ---------------- input index map (jax tree: sorted dict keys) -----------
enum {
  IN_ATOMS = 0, IN_ADJ, IN_WALK, IN_N1, IN_N2, IN_N3,
  P_D0_W, P_D0_B, P_D1_W, P_D1_B, P_D2_W, P_D2_B,
  P_DB1_W, P_DB1_B, P_DB2_W, P_DB2_B, P_DB3_W, P_DB3_B,
  P_DW1_W, P_DW1_B, P_DW2_W, P_DW2_B, P_DW3_W, P_DW3_B,
  P_GC_W, P_GC_B,
  P_GB_U, P_GB_W, P_GB_BH, P_GB_BI,
  P_GB0_U, P_GB0_W, P_GB0_BH, P_GB0_BI,
  P_GB1_U, P_GB1_W, P_GB1_BH, P_GB1_BI,
  P_GF_U, P_GF_W, P_GF_BH, P_GF_BI,
  P_GF0_U, P_GF0_W, P_GF0_BH, P_GF0_BI,
  P_GF1_U, P_GF1_W, P_GF1_BH, P_GF1_BI,
  P_GX_U, P_GX_W, P_GX_BH, P_GX_BI,
  P_GX0_U, P_GX0_W, P_GX0_BH, P_GX0_BI,
  P_GX1_U, P_GX1_W, P_GX1_BH, P_GX1_BI
};

extern "C" void kernel_launch(void* const* d_in, const int* in_sizes, int n_in,
                              void* d_out, int out_size, void* d_ws, size_t ws_size,
                              hipStream_t stream) {
  (void)in_sizes; (void)n_in; (void)out_size; (void)ws_size;
  char* ws = (char*)d_ws;
  size_t off = 0;
  auto alloc = [&](size_t bytes) -> void* {
    void* p = (void*)(ws + off);
    off += (bytes + 255) & ~(size_t)255;
    return p;
  };
  auto F = [&](int i) { return (const float*)d_in[i]; };

  // swizzled bf16 weights (K*N*2 bytes each)
  bf16* gf0U = (bf16*)alloc(128*384*2); bf16* gf0W = (bf16*)alloc(64*384*2);
  bf16* gf1U = (bf16*)alloc(128*384*2); bf16* gf1W = (bf16*)alloc(128*384*2);
  bf16* gfU  = (bf16*)alloc(128*384*2); bf16* gfW  = (bf16*)alloc(128*384*2);
  bf16* gb0U = (bf16*)alloc(128*384*2); bf16* gb0W = (bf16*)alloc(64*384*2);
  bf16* gb1U = (bf16*)alloc(128*384*2); bf16* gb1W = (bf16*)alloc(128*384*2);
  bf16* gbU  = (bf16*)alloc(128*384*2); bf16* gbW  = (bf16*)alloc(128*384*2);
  bf16* gx0U = (bf16*)alloc(128*384*2);
  bf16* gx1U = (bf16*)alloc(128*384*2); bf16* gx1W = (bf16*)alloc(128*384*2);
  bf16* gxU  = (bf16*)alloc(128*384*2); bf16* gxW  = (bf16*)alloc(128*384*2);
  bf16* d0Wz = (bf16*)alloc(704*64*2);
  // f32 activation buffers
  float* hw     = (float*)alloc(32*32*64*4);
  float* xw     = (float*)alloc(1024*384*4);
  float* bufA   = (float*)alloc(1024*128*4);
  float* bufB   = (float*)alloc(1024*128*4);
  float* fseq   = (float*)alloc(1024*128*4);
  float* bseq   = (float*)alloc(1024*128*4);
  float* fstate = (float*)alloc(32*128*4);
  float* bstate = (float*)alloc(32*128*4);

  auto swz = [&](int pi, bf16* dst, int K, int N) {
    int total = K * N;
    k_swizzle<<<(total + 255) / 256, 256, 0, stream>>>(F(pi), dst, K, N);
  };
  swz(P_GF0_U, gf0U, 128, 384); swz(P_GF0_W, gf0W, 64, 384);
  swz(P_GF1_U, gf1U, 128, 384); swz(P_GF1_W, gf1W, 128, 384);
  swz(P_GF_U,  gfU,  128, 384); swz(P_GF_W,  gfW,  128, 384);
  swz(P_GB0_U, gb0U, 128, 384); swz(P_GB0_W, gb0W, 64, 384);
  swz(P_GB1_U, gb1U, 128, 384); swz(P_GB1_W, gb1W, 128, 384);
  swz(P_GB_U,  gbU,  128, 384); swz(P_GB_W,  gbW,  128, 384);
  swz(P_GX0_U, gx0U, 128, 384);
  swz(P_GX1_U, gx1U, 128, 384); swz(P_GX1_W, gx1W, 128, 384);
  swz(P_GX_U,  gxU,  128, 384); swz(P_GX_W,  gxW,  128, 384);
  swz(P_D0_W,  d0Wz, 704, 64);

  k_graph<<<1, 256, 0, stream>>>(F(IN_ATOMS), F(IN_ADJ), (const int*)d_in[IN_WALK],
                                 F(P_GC_W), F(P_GC_B), hw);

  const size_t smemA = 16384 + 8192 + 49152 + 4096;
  k_gru_layer<<<1, 256, smemA, stream>>>(hw,   64,  gf0W, gf0U, F(P_GF0_BI), F(P_GF0_BH), xw, bufA, nullptr, 0);
  k_gru_layer<<<1, 256, smemA, stream>>>(bufA, 128, gf1W, gf1U, F(P_GF1_BI), F(P_GF1_BH), xw, bufB, nullptr, 0);
  k_gru_layer<<<1, 256, smemA, stream>>>(bufB, 128, gfW,  gfU,  F(P_GF_BI),  F(P_GF_BH),  xw, fseq, fstate, 0);
  k_gru_layer<<<1, 256, smemA, stream>>>(hw,   64,  gb0W, gb0U, F(P_GB0_BI), F(P_GB0_BH), xw, bufA, nullptr, 1);
  k_gru_layer<<<1, 256, smemA, stream>>>(bufA, 128, gb1W, gb1U, F(P_GB1_BI), F(P_GB1_BH), xw, bufB, nullptr, 1);
  k_gru_layer<<<1, 256, smemA, stream>>>(bufB, 128, gbW,  gbU,  F(P_GB_BI),  F(P_GB_BH),  xw, bseq, bstate, 1);

  ARArgs A;
  A.walk = (const int*)d_in[IN_WALK];
  A.noise2 = F(IN_N2); A.noise3 = F(IN_N3);
  A.hw = hw; A.fseq = fseq; A.bseq = bseq; A.fstate = fstate; A.bstate = bstate;
  A.W0raw = F(P_GX0_W); A.U0swz = gx0U; A.bi0 = F(P_GX0_BI); A.bh0 = F(P_GX0_BH);
  A.W1swz = gx1W; A.U1swz = gx1U; A.bi1 = F(P_GX1_BI); A.bh1 = F(P_GX1_BH);
  A.W2swz = gxW;  A.U2swz = gxU;  A.bi2 = F(P_GX_BI);  A.bh2 = F(P_GX_BH);
  A.d0Wswz = d0Wz; A.d0b = F(P_D0_B);
  A.d1W = F(P_D1_W); A.d1b = F(P_D1_B);
  A.d2W = F(P_D2_W); A.d2b = F(P_D2_B);
  A.dw1W = F(P_DW1_W); A.dw1b = F(P_DW1_B); A.db1W = F(P_DB1_W); A.db1b = F(P_DB1_B);
  A.dw2W = F(P_DW2_W); A.dw2b = F(P_DW2_B); A.db2W = F(P_DB2_W); A.db2b = F(P_DB2_B);
  A.dw3W = F(P_DW3_W); A.dw3b = F(P_DW3_B); A.db3W = F(P_DB3_W); A.db3b = F(P_DB3_B);
  A.out = (float*)d_out;
  k_autoreg<<<1, 256, AR_SMEM, stream>>>(A);
}
